// TransformerEncoderBlock_18528488914992
// MI455X (gfx1250) — matrix operand order repaired and checked by reading, not compiled, not run on hardware
//
#include <hip/hip_runtime.h>
#include <hip/hip_bf16.h>

// ---------------------------------------------------------------------------
// CDNA5 / gfx1250 transformer encoder block.
// bf16 WMMA (16x16x32, f32 accum) + async global->LDS staging (ASYNCcnt).
// ---------------------------------------------------------------------------

typedef __attribute__((ext_vector_type(16))) __bf16    v16bf;
typedef __attribute__((ext_vector_type(8)))  float     v8f;
typedef __attribute__((ext_vector_type(8)))  unsigned  v8u;

__device__ __forceinline__ unsigned short f2bf(float f) {
    unsigned b = __float_as_uint(f);
    b += 0x7FFFu + ((b >> 16) & 1u);          // round-to-nearest-even
    return (unsigned short)(b >> 16);
}
__device__ __forceinline__ unsigned pack2bf(float lo, float hi) {
    return ((unsigned)f2bf(hi) << 16) | (unsigned)f2bf(lo);
}
__device__ __forceinline__ v16bf as_v16bf(v8u u) {
    return __builtin_bit_cast(v16bf, u);
}
__device__ __forceinline__ v8f wmma_bf16(v16bf a, v16bf b, v8f c) {
    // (neg_a, A, neg_b, B, c_mod, C, reuse_a, reuse_b)
    return __builtin_amdgcn_wmma_f32_16x16x32_bf16(false, a, false, b, (short)0, c, false, false);
}

// A fragment: 16x32 bf16, row-major source. Lane l: row=l&15, half=l>>4.
// VGPR0..3 <- K = 8*half..+7 ; VGPR4..7 <- K = 16+8*half..+7 (ISA 7.12.2).
__device__ __forceinline__ v16bf load_a_bf16(const unsigned short* p, int stride) {
    int lane = threadIdx.x & 31;
    int row  = lane & 15, half = lane >> 4;
    const uint4* c0 = (const uint4*)(p + (size_t)row * stride + 8 * half);
    const uint4* c1 = (const uint4*)(p + (size_t)row * stride + 16 + 8 * half);
    uint4 a = *c0, b = *c1;
    v8u u; u[0]=a.x; u[1]=a.y; u[2]=a.z; u[3]=a.w; u[4]=b.x; u[5]=b.y; u[6]=b.z; u[7]=b.w;
    return as_v16bf(u);
}

// B fragment: 32x16 bf16, source stored transposed (N rows x K cols).
// Lane l: col=l&15, half=l>>4; VGPR0..3 <- K = 8*half..+7 ; VGPR4..7 <- K = 16+8*half..+7.
__device__ __forceinline__ v16bf load_b_bf16(const unsigned short* p, int stride) {
    int lane = threadIdx.x & 31;
    int col  = lane & 15, half = lane >> 4;
    const uint4* c0 = (const uint4*)(p + (size_t)col * stride + 8 * half);
    const uint4* c1 = (const uint4*)(p + (size_t)col * stride + 16 + 8 * half);
    uint4 a = *c0, b = *c1;
    v8u u; u[0]=a.x; u[1]=a.y; u[2]=a.z; u[3]=a.w; u[4]=b.x; u[5]=b.y; u[6]=b.z; u[7]=b.w;
    return as_v16bf(u);
}

// A fragment from f32 source (attention probabilities), convert to bf16.
__device__ __forceinline__ v16bf load_a_f32(const float* p, int stride) {
    int lane = threadIdx.x & 31;
    int row  = lane & 15, half = lane >> 4;
    const float4* c0 = (const float4*)(p + (size_t)row * stride + 8 * half);
    const float4* c1 = (const float4*)(p + (size_t)row * stride + 16 + 8 * half);
    float4 a0 = c0[0], a1 = c0[1], b0 = c1[0], b1 = c1[1];
    v8u u;
    u[0] = pack2bf(a0.x, a0.y); u[1] = pack2bf(a0.z, a0.w);
    u[2] = pack2bf(a1.x, a1.y); u[3] = pack2bf(a1.z, a1.w);
    u[4] = pack2bf(b0.x, b0.y); u[5] = pack2bf(b0.z, b0.w);
    u[6] = pack2bf(b1.x, b1.y); u[7] = pack2bf(b1.z, b1.w);
    return as_v16bf(u);
}

// ---------------------------------------------------------------------------
// Async global->LDS staging of one 128x32 bf16 tile (8KB), 256 threads.
// LDS layout: 128 rows, row stride 40 elems (80B) to spread banks.
// Each wave issues 2 async b128 instructions per tile (tracked on ASYNCcnt).
// ---------------------------------------------------------------------------
#define TILE_LDS_STRIDE 40   // elems; 80 bytes per row

__device__ __forceinline__ void stage_tile_async(
    const unsigned short* __restrict__ g, int ldg, unsigned lds_base, int tid)
{
#pragma unroll
    for (int i = 0; i < 2; i++) {
        int c   = tid + i * 256;          // 0..511 chunk id (16B chunks)
        int row = c >> 2;                 // 0..127
        int off = (c & 3) * 8;            // elem offset in row
        const unsigned short* gp = g + (size_t)row * ldg + off;
        unsigned la = lds_base + (unsigned)(row * (TILE_LDS_STRIDE * 2) + (c & 3) * 16);
        asm volatile("global_load_async_to_lds_b128 %0, %1, off"
                     :: "v"(la), "v"(gp) : "memory");
    }
}

// ---------------------------------------------------------------------------
// Generic WMMA GEMM: C[M,N] = A[M,K](bf16,row-major) @ BT[N,K](bf16)^T
// Block: 256 threads = 8 waves; block tile 128x128; wave tile 64x32 (4x2 WMMA).
// Double-buffered async LDS staging; fused bias/residual/ReLU/bf16 epilogue.
// ---------------------------------------------------------------------------
template <bool RELU, bool RESID, bool STORE_F, bool STORE_B>
__global__ __launch_bounds__(256) void gemm_bf16_kernel(
    const unsigned short* __restrict__ A, const unsigned short* __restrict__ BT,
    const float* __restrict__ bias, const float* __restrict__ resid,
    float* __restrict__ outF, unsigned short* __restrict__ outB,
    int M, int N, int K)
{
    __shared__ __align__(16) unsigned short Atile[2][128 * TILE_LDS_STRIDE];
    __shared__ __align__(16) unsigned short Btile[2][128 * TILE_LDS_STRIDE];

    int tid  = threadIdx.x;
    int wave = tid >> 5, lane = tid & 31;
    int wm = wave & 1, wn = wave >> 1;
    int m0 = blockIdx.y * 128;
    int n0 = blockIdx.x * 128;

    const unsigned short* Ablk = A  + (size_t)m0 * K;
    const unsigned short* Bblk = BT + (size_t)n0 * K;
    unsigned aBase[2] = { (unsigned)(uintptr_t)&Atile[0][0], (unsigned)(uintptr_t)&Atile[1][0] };
    unsigned bBase[2] = { (unsigned)(uintptr_t)&Btile[0][0], (unsigned)(uintptr_t)&Btile[1][0] };

    v8f acc[4][2];
#pragma unroll
    for (int i = 0; i < 4; i++)
#pragma unroll
        for (int j = 0; j < 2; j++)
            acc[i][j] = v8f{0.f,0.f,0.f,0.f,0.f,0.f,0.f,0.f};

    // prologue: stage k-step 0 into buffer 0 (4 async instr per wave)
    stage_tile_async(Ablk, K, aBase[0], tid);
    stage_tile_async(Bblk, K, bBase[0], tid);

    int nsteps = K >> 5;
    for (int s = 0; s < nsteps; s++) {
        int cur = s & 1;
        if (s + 1 < nsteps) {
            stage_tile_async(Ablk + (size_t)(s + 1) * 32, K, aBase[cur ^ 1], tid);
            stage_tile_async(Bblk + (size_t)(s + 1) * 32, K, bBase[cur ^ 1], tid);
            asm volatile("s_wait_asynccnt 0x4" ::: "memory");   // oldest stage done
        } else {
            asm volatile("s_wait_asynccnt 0x0" ::: "memory");
        }
        __syncthreads();

        v16bf a[4], b[2];
#pragma unroll
        for (int i = 0; i < 4; i++)
            a[i] = load_a_bf16(&Atile[cur][(wm * 64 + 16 * i) * TILE_LDS_STRIDE], TILE_LDS_STRIDE);
#pragma unroll
        for (int j = 0; j < 2; j++)
            b[j] = load_b_bf16(&Btile[cur][(wn * 32 + 16 * j) * TILE_LDS_STRIDE], TILE_LDS_STRIDE);
#pragma unroll
        for (int i = 0; i < 4; i++)
#pragma unroll
            for (int j = 0; j < 2; j++)
                acc[i][j] = wmma_bf16(a[i], b[j], acc[i][j]);

        __syncthreads();   // all waves done reading buf[cur] before it is restaged
    }

    int colb = lane & 15, halfr = lane >> 4;
#pragma unroll
    for (int j = 0; j < 2; j++) {
        int col = n0 + wn * 32 + 16 * j + colb;
        float bv = bias[col];
#pragma unroll
        for (int i = 0; i < 4; i++) {
#pragma unroll
            for (int r = 0; r < 8; r++) {
                int row = m0 + wm * 64 + 16 * i + r + 8 * halfr;
                size_t idx = (size_t)row * N + col;
                float val = acc[i][j][r] + bv;
                if (RESID) val += resid[idx];
                if (RELU)  val = fmaxf(val, 0.f);
                if (STORE_F) outF[idx] = val;
                if (STORE_B) outB[idx] = f2bf(val);
            }
        }
    }
}

// ---------------------------------------------------------------------------
// Attention raw scores: per block = one (b,h,16-row q tile); 8 waves split the
// 2048 key columns. Non-temporal stores (attn stream >> L2).
// ---------------------------------------------------------------------------
__global__ __launch_bounds__(256) void scores_kernel(
    const unsigned short* __restrict__ q, const unsigned short* __restrict__ k,
    const int* __restrict__ mask, float* __restrict__ attn, float scale)
{
    const int S = 2048, D = 1024, H = 16, DK = 64;
    int wave = threadIdx.x >> 5, lane = threadIdx.x & 31;
    int blk = blockIdx.x;
    int qt = blk & 127;          // S/16 = 128
    int hh = (blk >> 7) & 15;
    int bb = blk >> 11;

    const unsigned short* qbase = q + (size_t)(bb * S + qt * 16) * D + hh * DK;
    v16bf a0 = load_a_bf16(qbase + 0,  D);
    v16bf a1 = load_a_bf16(qbase + 32, D);

    int colb = lane & 15, halfr = lane >> 4;
    for (int t = 0; t < 16; t++) {
        int c0 = wave * 256 + t * 16;
        const unsigned short* kb = k + (size_t)(bb * S + c0) * D + hh * DK;
        v16bf b0 = load_b_bf16(kb + 0,  D);
        v16bf b1 = load_b_bf16(kb + 32, D);
        v8f acc = v8f{0.f,0.f,0.f,0.f,0.f,0.f,0.f,0.f};
        acc = wmma_bf16(a0, b0, acc);
        acc = wmma_bf16(a1, b1, acc);

        int col = c0 + colb;
        bool mz = (mask[bb * S + col] == 0);
        float* orow = attn + ((size_t)(bb * H + hh) * S + qt * 16) * S + col;
#pragma unroll
        for (int r = 0; r < 8; r++) {
            int roff = r + 8 * halfr;
            float v = mz ? -INFINITY : acc[r] * scale;
            __builtin_nontemporal_store(v, orow + (size_t)roff * S);
        }
    }
}

// Row softmax in place: one block per row of 2048 (non-temporal traffic).
__global__ __launch_bounds__(256) void softmax_kernel(float* __restrict__ attn) {
    __shared__ float red[256];
    float* rowp = attn + (size_t)blockIdx.x * 2048;
    int tid = threadIdx.x;
    float vals[8];
    float mx = -INFINITY;
#pragma unroll
    for (int i = 0; i < 8; i++) {
        vals[i] = __builtin_nontemporal_load(rowp + tid + i * 256);
        mx = fmaxf(mx, vals[i]);
    }
    red[tid] = mx; __syncthreads();
    for (int s = 128; s > 0; s >>= 1) { if (tid < s) red[tid] = fmaxf(red[tid], red[tid + s]); __syncthreads(); }
    mx = red[0]; __syncthreads();
    float sum = 0.f;
#pragma unroll
    for (int i = 0; i < 8; i++) { vals[i] = __expf(vals[i] - mx); sum += vals[i]; }
    red[tid] = sum; __syncthreads();
    for (int s = 128; s > 0; s >>= 1) { if (tid < s) red[tid] += red[tid + s]; __syncthreads(); }
    float inv = 1.0f / red[0];
#pragma unroll
    for (int i = 0; i < 8; i++)
        __builtin_nontemporal_store(vals[i] * inv, rowp + tid + i * 256);
}

// v (tokens,D) bf16 -> vT (B,H,DK,S) bf16 so attn@V B-fragments are contiguous.
__global__ __launch_bounds__(256) void transpose_v_kernel(
    const unsigned short* __restrict__ v, unsigned short* __restrict__ vT)
{
    size_t o = (size_t)blockIdx.x * 256 + threadIdx.x;   // B*H*64*S elements
    int s  = (int)(o & 2047);
    size_t rest = o >> 11;
    int d  = (int)(rest & 63);
    int bh = (int)(rest >> 6);
    int hh = bh & 15, bb = bh >> 4;
    vT[o] = v[(size_t)(bb * 2048 + s) * 1024 + hh * 64 + d];
}

// context = attn @ V. Block = 128 q-rows of one (b,h); wave = 16 rows x 64 cols.
__global__ __launch_bounds__(256) void attn_v_kernel(
    const float* __restrict__ attn, const unsigned short* __restrict__ vT,
    unsigned short* __restrict__ ctx)
{
    const int S = 2048, D = 1024, H = 16, DK = 64;
    int wave = threadIdx.x >> 5, lane = threadIdx.x & 31;
    int blk = blockIdx.x;               // B*H*(S/128) = 512
    int mt = blk & 15;
    int hh = (blk >> 4) & 15;
    int bb = blk >> 8;
    int m0 = mt * 128 + wave * 16;

    const float*          arow = attn + ((size_t)(bb * H + hh) * S + m0) * S;
    const unsigned short* vb   = vT + (size_t)(bb * H + hh) * DK * S;

    v8f acc[4];
#pragma unroll
    for (int j = 0; j < 4; j++) acc[j] = v8f{0.f,0.f,0.f,0.f,0.f,0.f,0.f,0.f};

    for (int k0 = 0; k0 < S; k0 += 32) {
        v16bf a = load_a_f32(arow + k0, S);
#pragma unroll
        for (int j = 0; j < 4; j++) {
            v16bf b = load_b_bf16(vb + (size_t)(16 * j) * S + k0, S);
            acc[j] = wmma_bf16(a, b, acc[j]);
        }
    }

    int colb = lane & 15, halfr = lane >> 4;
#pragma unroll
    for (int j = 0; j < 4; j++) {
        int d = 16 * j + colb;
#pragma unroll
        for (int r = 0; r < 8; r++) {
            int row = m0 + r + 8 * halfr;
            ctx[(size_t)(bb * S + row) * D + hh * DK + d] = f2bf(acc[j][r]);
        }
    }
}

// rmsnorm over D=1024; one block per token; optional f32 and bf16 outputs.
__global__ __launch_bounds__(256) void rmsnorm_kernel(
    const float* __restrict__ in, const float* __restrict__ g,
    float* __restrict__ outF, unsigned short* __restrict__ outB)
{
    const int D = 1024;
    __shared__ float red[256];
    const float* row = in + (size_t)blockIdx.x * D;
    int tid = threadIdx.x;
    float v[4]; float ss = 0.f;
#pragma unroll
    for (int i = 0; i < 4; i++) { v[i] = row[tid + i * 256]; ss += v[i] * v[i]; }
    red[tid] = ss; __syncthreads();
    for (int s = 128; s > 0; s >>= 1) { if (tid < s) red[tid] += red[tid + s]; __syncthreads(); }
    float rs = rsqrtf(red[0] * (1.0f / D) + 1e-8f);
#pragma unroll
    for (int i = 0; i < 4; i++) {
        int c = tid + i * 256;
        float val = v[i] * rs * g[c];
        if (outF) outF[(size_t)blockIdx.x * D + c] = val;
        if (outB) outB[(size_t)blockIdx.x * D + c] = f2bf(val);
    }
}

__global__ __launch_bounds__(256) void f32_to_bf16_kernel(
    const float* __restrict__ in, unsigned short* __restrict__ out, int n)
{
    int i = blockIdx.x * 256 + threadIdx.x;
    if (i < n) out[i] = f2bf(in[i]);
}

// W (K,N) f32 row-major -> WT (N,K) bf16 (coalesced writes).
__global__ __launch_bounds__(256) void transpose_w_kernel(
    const float* __restrict__ W, unsigned short* __restrict__ WT, int K, int N)
{
    size_t o = (size_t)blockIdx.x * 256 + threadIdx.x;
    if (o < (size_t)K * N) {
        int n  = (int)(o / K);
        int kk = (int)(o % K);
        WT[o] = f2bf(W[(size_t)kk * N + n]);
    }
}

// ---------------------------------------------------------------------------
extern "C" void kernel_launch(void* const* d_in, const int* in_sizes, int n_in,
                              void* d_out, int out_size, void* d_ws, size_t ws_size,
                              hipStream_t stream) {
    const int B = 2, S = 2048, D = 1024, F = 4096, H = 16, DK = 64;
    const int MT = B * S;                       // 4096 tokens

    const float* x    = (const float*)d_in[0];
    const int*   mask = (const int*)d_in[1];
    const float* Wq = (const float*)d_in[2];  const float* bq = (const float*)d_in[3];
    const float* Wk = (const float*)d_in[4];  const float* bk = (const float*)d_in[5];
    const float* Wv = (const float*)d_in[6];  const float* bv = (const float*)d_in[7];
    const float* Wo = (const float*)d_in[8];  const float* bo = (const float*)d_in[9];
    const float* g1 = (const float*)d_in[10];
    const float* W1 = (const float*)d_in[11]; const float* b1 = (const float*)d_in[12];
    const float* W2 = (const float*)d_in[13]; const float* b2 = (const float*)d_in[14];
    const float* g2 = (const float*)d_in[15];

    float* y    = (float*)d_out;                          // (B,S,D)
    float* attn = y + (size_t)MT * D;                     // (B,H,S,S)

    char* wp = (char*)d_ws;
    auto alloc = [&](size_t bytes) -> char* {
        char* p = wp; wp += (bytes + 255) & ~(size_t)255; return p;
    };
    unsigned short* xb   = (unsigned short*)alloc((size_t)MT * D * 2);
    unsigned short* WqT  = (unsigned short*)alloc((size_t)D * D * 2);
    unsigned short* WkT  = (unsigned short*)alloc((size_t)D * D * 2);
    unsigned short* WvT  = (unsigned short*)alloc((size_t)D * D * 2);
    unsigned short* WoT  = (unsigned short*)alloc((size_t)D * D * 2);
    unsigned short* W1T  = (unsigned short*)alloc((size_t)D * F * 2);
    unsigned short* W2T  = (unsigned short*)alloc((size_t)F * D * 2);
    unsigned short* qb   = (unsigned short*)alloc((size_t)MT * D * 2);
    unsigned short* kb   = (unsigned short*)alloc((size_t)MT * D * 2);
    unsigned short* vb   = (unsigned short*)alloc((size_t)MT * D * 2);
    unsigned short* vTb  = (unsigned short*)alloc((size_t)B * H * DK * S * 2);
    unsigned short* ctx  = (unsigned short*)alloc((size_t)MT * D * 2);
    float*          hpre = (float*)alloc((size_t)MT * D * 4);   // reused for pre-y
    float*          hf   = (float*)alloc((size_t)MT * D * 4);
    unsigned short* hb   = (unsigned short*)alloc((size_t)MT * D * 2);
    unsigned short* ffn  = (unsigned short*)alloc((size_t)MT * F * 2);

    // ---- prep: bf16 conversions / weight transposes ----
    f32_to_bf16_kernel<<<(MT * D) / 256, 256, 0, stream>>>(x, xb, MT * D);
    transpose_w_kernel<<<(D * D) / 256, 256, 0, stream>>>(Wq, WqT, D, D);
    transpose_w_kernel<<<(D * D) / 256, 256, 0, stream>>>(Wk, WkT, D, D);
    transpose_w_kernel<<<(D * D) / 256, 256, 0, stream>>>(Wv, WvT, D, D);
    transpose_w_kernel<<<(D * D) / 256, 256, 0, stream>>>(Wo, WoT, D, D);
    transpose_w_kernel<<<(D * F) / 256, 256, 0, stream>>>(W1, W1T, D, F);
    transpose_w_kernel<<<(F * D) / 256, 256, 0, stream>>>(W2, W2T, F, D);

    // ---- QKV projections ----
    dim3 gQ(D / 128, MT / 128);
    gemm_bf16_kernel<false, false, false, true><<<gQ, 256, 0, stream>>>(
        xb, WqT, bq, nullptr, nullptr, qb, MT, D, D);
    gemm_bf16_kernel<false, false, false, true><<<gQ, 256, 0, stream>>>(
        xb, WkT, bk, nullptr, nullptr, kb, MT, D, D);
    gemm_bf16_kernel<false, false, false, true><<<gQ, 256, 0, stream>>>(
        xb, WvT, bv, nullptr, nullptr, vb, MT, D, D);

    // ---- attention ----
    scores_kernel<<<B * H * (S / 16), 256, 0, stream>>>(qb, kb, mask, attn, 0.125f);
    softmax_kernel<<<B * H * S, 256, 0, stream>>>(attn);
    transpose_v_kernel<<<(B * H * DK * S) / 256, 256, 0, stream>>>(vb, vTb);
    attn_v_kernel<<<B * H * (S / 128), 256, 0, stream>>>(attn, vTb, ctx);

    // ---- output projection + residual, rmsnorm ----
    gemm_bf16_kernel<false, true, true, false><<<gQ, 256, 0, stream>>>(
        ctx, WoT, bo, x, hpre, nullptr, MT, D, D);
    rmsnorm_kernel<<<MT, 256, 0, stream>>>(hpre, g1, hf, hb);

    // ---- FFN ----
    dim3 gF1(F / 128, MT / 128);
    gemm_bf16_kernel<true, false, false, true><<<gF1, 256, 0, stream>>>(
        hb, W1T, b1, nullptr, nullptr, ffn, MT, F, D);
    dim3 gF2(D / 128, MT / 128);
    gemm_bf16_kernel<false, true, true, false><<<gF2, 256, 0, stream>>>(
        ffn, W2T, b2, hf, hpre, nullptr, MT, D, F);
    rmsnorm_kernel<<<MT, 256, 0, stream>>>(hpre, g2, y, nullptr);

    (void)in_sizes; (void)n_in; (void)out_size; (void)ws_size;
}
